// MultiFeatureMOE_34772055229069
// MI455X (gfx1250) — matrix operand order repaired and checked by reading, not compiled, not run on hardware
//
#include <hip/hip_runtime.h>
#include <hip/hip_bf16.h>

#define F_ 4
#define E_ 8
#define B_ 4096
#define D_ 512
#define H_ 1024
#define O_ 512

typedef __attribute__((ext_vector_type(16))) __bf16 v16bf;
typedef __attribute__((ext_vector_type(8)))  __bf16 v8bf;
typedef __attribute__((ext_vector_type(8)))  float  v8f;

union Frag { v16bf v; v8bf h[2]; };

// f32 -> bf16 round-to-nearest-even via integer math
__device__ __forceinline__ unsigned short f2bf(float x) {
  union { float f; unsigned int u; } c; c.f = x;
  unsigned int u = c.u;
  unsigned int r = u + 0x7FFFu + ((u >> 16) & 1u);
  return (unsigned short)(r >> 16);
}
__device__ __forceinline__ unsigned int pack_bf2(float a, float b) {
  return (unsigned int)f2bf(a) | ((unsigned int)f2bf(b) << 16);
}

__device__ __forceinline__ v8f wmma_bf16(const Frag& a, const Frag& b, v8f c) {
  // D = A(16x32 bf16) * B(32x16 bf16) + C(16x16 f32)
  return __builtin_amdgcn_wmma_f32_16x16x32_bf16(false, a.v, false, b.v,
                                                 (short)0, c, false, false);
}

// ---- CDNA5 async global->LDS copy (ASYNCcnt-tracked DMA, GVS addressing) ----
// LDS dest address per lane from the VDST VGPR; global address = SGPR base + per-lane VGPR offset
__device__ __forceinline__ unsigned lds_addr_of(const void* p) {
  // generic pointers to LDS carry the DS offset in the low 32 bits (ISA 10.2)
  return (unsigned)(unsigned long long)p;
}
__device__ __forceinline__ void async_load_b128(unsigned lds, unsigned voff,
                                                const void* sbase) {
  asm volatile("global_load_async_to_lds_b128 %0, %1, %2"
               :: "v"(lds), "v"(voff), "s"(sbase)
               : "memory");
}
__device__ __forceinline__ void wait_async0() {
  asm volatile("s_wait_asynccnt 0x0" ::: "memory");
}

// ---------------------------------------------------------------------------
__global__ void zero_kernel(float* __restrict__ p, int n) {
  int i = blockIdx.x * blockDim.x + threadIdx.x;
  if (i < n) p[i] = 0.f;
}

// elementwise f32 -> bf16 (n4 = n/4 float4 groups)
__global__ __launch_bounds__(256)
void conv_kernel(const float* __restrict__ in, unsigned short* __restrict__ outp, int n4) {
  int i = blockIdx.x * 256 + threadIdx.x;
  if (i < n4) {
    float4 x = ((const float4*)in)[i];
    uint2 r;
    r.x = pack_bf2(x.x, x.y);
    r.y = pack_bf2(x.z, x.w);
    ((uint2*)outp)[i] = r;
  }
}

// tiled transpose + convert: in[z][R][C] f32  ->  out[z][C][R] bf16
__global__ __launch_bounds__(256)
void tconv_kernel(const float* __restrict__ in, unsigned short* __restrict__ outp,
                  int R, int C) {
  __shared__ float tile[32][33];
  const float* src = in + (size_t)blockIdx.z * R * C;
  unsigned short* dst = outp + (size_t)blockIdx.z * R * C;
  const int c0 = blockIdx.x * 32, r0 = blockIdx.y * 32;
  const int tx = threadIdx.x, ty = threadIdx.y;
#pragma unroll
  for (int i = ty; i < 32; i += 8)
    tile[i][tx] = src[(size_t)(r0 + i) * C + c0 + tx];
  __syncthreads();
#pragma unroll
  for (int i = ty; i < 32; i += 8)
    dst[(size_t)(c0 + i) * R + r0 + tx] = f2bf(tile[tx][i]);
}

// ---------------------------------------------------------------------------
// gate[f,b,e] = softmax_e( features[f,b,:] . Wg[f,:,e] + bg[f,e] )
__global__ __launch_bounds__(256)
void gate_kernel(const float* __restrict__ feat, const float* __restrict__ Wg,
                 const float* __restrict__ bg, float* __restrict__ gate) {
  __shared__ float Wgs[D_ * E_];                 // 16 KB, [d][e]
  const int f = blockIdx.x >> 4;                 // B/256 = 16 blocks per feature
  const int b = ((blockIdx.x & 15) << 8) + threadIdx.x;
  const float* wgf = Wg + (size_t)f * D_ * E_;
  for (int i = threadIdx.x; i < D_ * E_; i += 256) Wgs[i] = wgf[i];
  __syncthreads();

  float acc[E_];
#pragma unroll
  for (int e = 0; e < E_; ++e) acc[e] = bg[f * E_ + e];

  const float* xp = feat + (size_t)(f * B_ + b) * D_;
  for (int d = 0; d < D_; d += 4) {
    float4 x = *(const float4*)(xp + d);
#pragma unroll
    for (int e = 0; e < E_; ++e) {
      acc[e] += x.x * Wgs[(d + 0) * E_ + e]
             +  x.y * Wgs[(d + 1) * E_ + e]
             +  x.z * Wgs[(d + 2) * E_ + e]
             +  x.w * Wgs[(d + 3) * E_ + e];
    }
  }
  float m = acc[0];
#pragma unroll
  for (int e = 1; e < E_; ++e) m = fmaxf(m, acc[e]);
  float s = 0.f;
#pragma unroll
  for (int e = 0; e < E_; ++e) { acc[e] = __expf(acc[e] - m); s += acc[e]; }
  float inv = 1.f / s;
  float* gp = gate + (size_t)(f * B_ + b) * E_;
#pragma unroll
  for (int e = 0; e < E_; ++e) gp[e] = acc[e] * inv;
}

// ---------------------------------------------------------------------------
// Fused 2-layer MoE MLP. One block = 32 batch rows x full O, loops experts.
constexpr int BT  = 32;        // batch tile
constexpr int HC  = 64;        // hidden chunk
constexpr int XP  = D_ + 8;    // 520: row pad -> 16 lanes stripe all 64 banks
constexpr int W1P = D_ + 8;    // 520
constexpr int W2P = 32 + 8;    // 40 (80B row stride, multiple of 16B)
constexpr int HP  = HC + 8;    // 72

// FAST: sources are pre-converted bf16 (Xbf row-major, W1bfT [h][d], W2bfT [o][h]);
//       staging uses global_load_async_to_lds_b128 overlapped with WMMA compute.
template <bool FAST>
__global__ __launch_bounds__(256)
void moe_kernel(const float* __restrict__ feat, const float* __restrict__ W1,
                const float* __restrict__ b1,  const float* __restrict__ W2,
                const float* __restrict__ b2,  const float* __restrict__ gate,
                const unsigned short* __restrict__ Xbf,
                const unsigned short* __restrict__ W1bfT,
                const unsigned short* __restrict__ W2bfT,
                float* __restrict__ out) {
  __shared__ __align__(16) unsigned short Xs[BT][XP];     // bf16 X tile
  __shared__ __align__(16) unsigned short W1sT[HC][W1P];  // [h_local][d]
  __shared__ __align__(16) unsigned short W2sT[O_][W2P];  // [o][k_local 0..31]
  __shared__ __align__(16) unsigned short Hs[BT][HP];     // gate-scaled relu(h)
  __shared__ float Gs[BT][E_];

  const int tid  = threadIdx.x;
  const int wave = tid >> 5;
  const int lane = tid & 31;
  const int l15  = lane & 15;
  const int lh   = lane >> 4;          // 0: lanes 0-15, 1: lanes 16-31
  const int f    = blockIdx.z;
  const int b0   = blockIdx.x * BT;

  const unsigned ldsX  = lds_addr_of(&Xs[0][0]);
  const unsigned ldsW1 = lds_addr_of(&W1sT[0][0]);
  const unsigned ldsW2 = lds_addr_of(&W2sT[0][0]);

  // ---- stage X tile (bf16) and gate rows, once per block ----
  if constexpr (FAST) {
    const unsigned short* xsrc = Xbf + (size_t)(f * B_ + b0) * D_;
    for (int i = tid * 8; i < BT * D_; i += 256 * 8) {     // 8 iterations
      unsigned r = (unsigned)i >> 9, d = (unsigned)i & 511u;
      async_load_b128(ldsX + r * (XP * 2) + d * 2, (unsigned)i * 2, xsrc);
    }
  } else {
    const float* xf = feat + (size_t)(f * B_ + b0) * D_;
    for (int i = tid; i < BT * D_; i += 256) {
      int r = i >> 9, d = i & 511;
      Xs[r][d] = f2bf(xf[(size_t)r * D_ + d]);
    }
  }
  {
    int r = tid >> 3, e = tid & 7;     // exactly 256 = 32*8 elements
    Gs[r][e] = gate[(size_t)(f * B_ + b0 + r) * E_ + e];
  }

  const int mt = wave >> 2;            // 0..1 -> rows m0..m0+15
  const int nq = wave & 3;             // GEMM1 n-tile / GEMM2 128-col group
  const int m0 = mt * 16;

  v8f acc2[8];                         // 8 output tiles per wave (128 of O=512 cols)
#pragma unroll
  for (int j = 0; j < 8; ++j)
#pragma unroll
    for (int r = 0; r < 8; ++r) acc2[j][r] = 0.f;

  for (int e = 0; e < E_; ++e) {
    const float* w1p = W1 + (size_t)(f * E_ + e) * D_ * H_;
    const float* w2p = W2 + (size_t)(f * E_ + e) * H_ * O_;
    const float* b1p = b1 + (size_t)(f * E_ + e) * H_;
    const unsigned short* w2tsrc =
        FAST ? (W2bfT + (size_t)(f * E_ + e) * O_ * H_) : nullptr;

    for (int hc = 0; hc < H_ / HC; ++hc) {
      // ---- stage W1 chunk as [h_local][d] (contiguous 64KB in W1bfT) ----
      if constexpr (FAST) {
        const unsigned short* s =
            W1bfT + ((size_t)(f * E_ + e) * H_ + (size_t)hc * HC) * D_;
        for (int i = tid * 8; i < HC * D_; i += 256 * 8) { // 16 iterations
          unsigned hl = (unsigned)i >> 9, d = (unsigned)i & 511u;
          async_load_b128(ldsW1 + hl * (W1P * 2) + d * 2, (unsigned)i * 2, s);
        }
        wait_async0();
      } else {
        for (int i = tid; i < HC * D_; i += 256) {
          int d = i >> 6, hl = i & 63;
          W1sT[hl][d] = f2bf(w1p[(size_t)d * H_ + hc * HC + hl]);
        }
      }
      __syncthreads();                 // W1sT (and X tile) ready

      // ---- issue W2 half 0 (k = 0..31): overlaps GEMM1 compute ----
      if constexpr (FAST) {
        const unsigned short* s = w2tsrc + hc * HC;
        for (int i = tid * 8; i < O_ * 32; i += 256 * 8) { // 8 iterations
          unsigned o = (unsigned)i >> 5, k = (unsigned)i & 31u;
          async_load_b128(ldsW2 + o * (W2P * 2) + k * 2,
                          (o * (unsigned)H_ + k) * 2, s);
        }
      } else {
        for (int i = tid; i < 32 * O_; i += 256) {
          int k = i >> 9, o = i & 511;
          W2sT[o][k] = f2bf(w2p[(size_t)(hc * HC + k) * O_ + o]);
        }
      }

      // ---- GEMM1: one 16x16 tile/wave, K = D = 512 in 16 WMMA steps ----
      v8f acc1;
      {
        float b1v = b1p[hc * HC + nq * 16 + l15];   // bias by output column
#pragma unroll
        for (int r = 0; r < 8; ++r) acc1[r] = b1v;
      }
#pragma unroll 4
      for (int kk = 0; kk < D_ / 32; ++kk) {
        Frag a, bf;
        // A (16x32 bf16): lane = row m0+l15, K = t + 16*lh
        a.h[0] = *(const v8bf*)&Xs[m0 + l15][kk * 32 + 16 * lh];
        a.h[1] = *(const v8bf*)&Xs[m0 + l15][kk * 32 + 16 * lh + 8];
        // B (32x16 bf16): lane = col nq*16+l15, K = t + 16*lh
        bf.h[0] = *(const v8bf*)&W1sT[nq * 16 + l15][kk * 32 + 16 * lh];
        bf.h[1] = *(const v8bf*)&W1sT[nq * 16 + l15][kk * 32 + 16 * lh + 8];
        acc1 = wmma_bf16(a, bf, acc1);
      }
      // relu, fold gate weight into rows, store bf16 hidden tile
#pragma unroll
      for (int r = 0; r < 8; ++r) {
        int row = m0 + r + 8 * lh;     // C/D layout: VGPR r -> M = r + 8*laneHi
        float v = acc1[r];
        v = v > 0.f ? v : 0.f;
        Hs[row][nq * 16 + l15] = f2bf(v * Gs[row][e]);
      }
      if constexpr (FAST) wait_async0();
      __syncthreads();                 // Hs stores + W2 half0 ready

      // ---- GEMM2 k-step 0 (hidden cols 0..31) ----
      {
        Frag a;
        a.h[0] = *(const v8bf*)&Hs[m0 + l15][0 + 16 * lh];
        a.h[1] = *(const v8bf*)&Hs[m0 + l15][8 + 16 * lh];
#pragma unroll
        for (int j = 0; j < 8; ++j) {
          Frag bf;
          int n = nq * 128 + j * 16 + l15;
          bf.h[0] = *(const v8bf*)&W2sT[n][16 * lh];
          bf.h[1] = *(const v8bf*)&W2sT[n][16 * lh + 8];
          acc2[j] = wmma_bf16(a, bf, acc2[j]);
        }
      }
      __syncthreads();                 // all waves done reading W2 half0

      // ---- stage W2 half 1 (k = 32..63) ----
      if constexpr (FAST) {
        const unsigned short* s = w2tsrc + hc * HC + 32;
        for (int i = tid * 8; i < O_ * 32; i += 256 * 8) {
          unsigned o = (unsigned)i >> 5, k = (unsigned)i & 31u;
          async_load_b128(ldsW2 + o * (W2P * 2) + k * 2,
                          (o * (unsigned)H_ + k) * 2, s);
        }
        wait_async0();
      } else {
        for (int i = tid; i < 32 * O_; i += 256) {
          int k = i >> 9, o = i & 511;
          W2sT[o][k] = f2bf(w2p[(size_t)(hc * HC + 32 + k) * O_ + o]);
        }
      }
      __syncthreads();                 // W2 half1 ready

      // ---- GEMM2 k-step 1 (hidden cols 32..63) ----
      {
        Frag a;
        a.h[0] = *(const v8bf*)&Hs[m0 + l15][32 + 16 * lh];
        a.h[1] = *(const v8bf*)&Hs[m0 + l15][40 + 16 * lh];
#pragma unroll
        for (int j = 0; j < 8; ++j) {
          Frag bf;
          int n = nq * 128 + j * 16 + l15;
          bf.h[0] = *(const v8bf*)&W2sT[n][16 * lh];
          bf.h[1] = *(const v8bf*)&W2sT[n][16 * lh + 8];
          acc2[j] = wmma_bf16(a, bf, acc2[j]);
        }
      }
      // no trailing barrier: next chunk's wait+sync orders all reuse hazards
    }
  }

  // ---- epilogue: stage b2[f] slice into dead W1sT space, then
  //      out += (acc2 + sum_e g*b2) / F with one atomic per element ----
  __syncthreads();                     // all WMMA reads of W1sT finished
  float* B2s = (float*)&W1sT[0][0];    // 8*512 f32 = 16KB reuse
  for (int i = tid; i < E_ * O_; i += 256)
    B2s[i] = b2[(size_t)f * E_ * O_ + i];
  __syncthreads();

#pragma unroll
  for (int j = 0; j < 8; ++j) {
    int o = nq * 128 + j * 16 + l15;
#pragma unroll
    for (int r = 0; r < 8; ++r) {
      int row = m0 + r + 8 * lh;
      float bias = 0.f;
#pragma unroll
      for (int e = 0; e < E_; ++e)
        bias += Gs[row][e] * B2s[e * O_ + o];
      float val = 0.25f * (acc2[j][r] + bias);   // 1/F
      atomicAdd(&out[(size_t)(b0 + row) * O_ + o], val);
    }
  }
}

// ---------------------------------------------------------------------------
extern "C" void kernel_launch(void* const* d_in, const int* in_sizes, int n_in,
                              void* d_out, int out_size, void* d_ws, size_t ws_size,
                              hipStream_t stream) {
  const float* feat = (const float*)d_in[0];   // [F,B,D]
  const float* W1   = (const float*)d_in[1];   // [F,E,D,H]
  const float* b1   = (const float*)d_in[2];   // [F,E,H]
  const float* W2   = (const float*)d_in[3];   // [F,E,H,O]
  const float* b2   = (const float*)d_in[4];   // [F,E,O]
  const float* Wg   = (const float*)d_in[5];   // [F,D,E]
  const float* bg   = (const float*)d_in[6];   // [F,E]
  float* out = (float*)d_out;                  // [B,O]

  // workspace layout
  const size_t gateB = (size_t)F_ * B_ * E_ * sizeof(float);        //   0.5 MB
  const size_t xB    = (size_t)F_ * B_ * D_ * 2;                    //  16.8 MB
  const size_t w1B   = (size_t)F_ * E_ * D_ * H_ * 2;               //  33.6 MB
  const size_t w2B   = (size_t)F_ * E_ * H_ * O_ * 2;               //  33.6 MB
  const size_t need  = gateB + xB + w1B + w2B;

  char* wsb = (char*)d_ws;
  float* gate_ws          = (float*)wsb;
  unsigned short* Xbf     = (unsigned short*)(wsb + gateB);
  unsigned short* W1bfT   = (unsigned short*)(wsb + gateB + xB);
  unsigned short* W2bfT   = (unsigned short*)(wsb + gateB + xB + w1B);

  const bool fast = (ws_size >= need);

  // out is accumulated atomically over F -> zero it first
  zero_kernel<<<(B_ * O_ + 255) / 256, 256, 0, stream>>>(out, B_ * O_);

  // gate softmax
  gate_kernel<<<F_ * B_ / 256, 256, 0, stream>>>(feat, Wg, bg, gate_ws);

  dim3 grid(B_ / BT, 1, F_);
  if (fast) {
    // pre-convert features, pre-transpose+convert weights (bf16)
    conv_kernel<<<(F_ * B_ * D_ / 4 + 255) / 256, 256, 0, stream>>>(
        feat, Xbf, F_ * B_ * D_ / 4);
    tconv_kernel<<<dim3(H_ / 32, D_ / 32, F_ * E_), dim3(32, 8), 0, stream>>>(
        W1, W1bfT, D_, H_);   // [d][h] -> [h][d]
    tconv_kernel<<<dim3(O_ / 32, H_ / 32, F_ * E_), dim3(32, 8), 0, stream>>>(
        W2, W2bfT, H_, O_);   // [h][o] -> [o][h]
    moe_kernel<true><<<grid, 256, 0, stream>>>(feat, W1, b1, W2, b2, gate_ws,
                                               Xbf, W1bfT, W2bfT, out);
  } else {
    moe_kernel<false><<<grid, 256, 0, stream>>>(feat, W1, b1, W2, b2, gate_ws,
                                                nullptr, nullptr, nullptr, out);
  }
}
